// KeypointStreamEncoder_3384434229465
// MI455X (gfx1250) — compile-verified, matrix operand order checked by reading, not run on hardware
//
#include <hip/hip_runtime.h>
#include <math.h>

// ---------------- problem constants ----------------
constexpr int BATCH = 8;
constexpr int L     = 1024;
constexpr int E     = 512;
constexpr int H     = 8;
constexpr int D     = 64;          // head dim
constexpr int BH    = BATCH * H;   // 64
constexpr int MROWS = BATCH * L;   // 8192
constexpr float SCALE = 0.125f;    // 1/sqrt(64)

// ---------------- CDNA5 WMMA plumbing ----------------
typedef __attribute__((ext_vector_type(16))) __bf16         v16bf;
typedef __attribute__((ext_vector_type(16))) unsigned short v16u;
typedef __attribute__((ext_vector_type(8)))  unsigned short u16x8;
typedef __attribute__((ext_vector_type(4)))  unsigned short u16x4;
typedef __attribute__((ext_vector_type(4)))  float          f32x4;
typedef __attribute__((ext_vector_type(8)))  float          v8f;

__device__ __forceinline__ unsigned short f2bf(float f) {
  union { float f; unsigned u; } c; c.f = f;
  unsigned u = c.u;
  u += 0x7FFFu + ((u >> 16) & 1u);   // round-to-nearest-even
  return (unsigned short)(u >> 16);
}
__device__ __forceinline__ float bf2f(unsigned short h) {
  union { unsigned u; float f; } c; c.u = ((unsigned)h) << 16;
  return c.f;
}

// D = A(16x32 bf16) * B(32x16 bf16) + C(16x16 f32), wave32
__device__ __forceinline__ v8f wmma_bf16(v16u a, v16u b, v8f c) {
  return __builtin_amdgcn_wmma_f32_16x16x32_bf16(
      false, __builtin_bit_cast(v16bf, a),
      false, __builtin_bit_cast(v16bf, b),
      (short)0, c, false, false);
}

// Fragment load: lane's 16 bf16 elements are two contiguous runs of 8:
//   k = half*8 + {0..7}  and  k = 16 + half*8 + {0..7}
// rowptr = base of this lane's row/col at the current 32-wide K window.
// Two b128 loads; rowptr + half*8 must be 16B aligned (all our layouts are).
__device__ __forceinline__ v16u load_frag(const unsigned short* rowptr, int half) {
  u16x8 lo = *(const u16x8*)(rowptr + half * 8);
  u16x8 hi = *(const u16x8*)(rowptr + 16 + half * 8);
  return __builtin_shufflevector(lo, hi, 0, 1, 2, 3, 4, 5, 6, 7,
                                 8, 9, 10, 11, 12, 13, 14, 15);
}

// ---------------- kernel 0: prep ----------------
// x -> bf16 (vectorized), weights -> bf16, mask -> float, zero stats.
__global__ __launch_bounds__(256) void prep_kernel(
    const float* __restrict__ x, unsigned short* __restrict__ xb,
    const float* __restrict__ Wq, const float* __restrict__ Wk,
    const float* __restrict__ Wv, const float* __restrict__ Wo,
    unsigned short* __restrict__ Wb,    // [4][E*E] bf16: q,k,v,o
    const int* __restrict__ mask, float* __restrict__ maskf,
    float* __restrict__ sumE, float* __restrict__ gsum) {
  int i = blockIdx.x * 256 + threadIdx.x;
  const int XN4 = MROWS * E / 4;   // 1048576 groups of 4
  const int WN4 = E * E / 4;       // 65536 groups of 4
  if (i < XN4) {
    f32x4 v = ((const f32x4*)x)[i];
    u16x4 o; o.x = f2bf(v.x); o.y = f2bf(v.y); o.z = f2bf(v.z); o.w = f2bf(v.w);
    ((u16x4*)xb)[i] = o;
  }
  if (i < WN4) {
    const float* src[4] = {Wq, Wk, Wv, Wo};
#pragma unroll
    for (int m = 0; m < 4; ++m) {
      f32x4 v = ((const f32x4*)src[m])[i];
      u16x4 o; o.x = f2bf(v.x); o.y = f2bf(v.y); o.z = f2bf(v.z); o.w = f2bf(v.w);
      ((u16x4*)(Wb + (size_t)m * E * E))[i] = o;
    }
  }
  if (i < BATCH * L) maskf[i] = (mask[i] != 0) ? 1.0f : 0.0f;
  if (i < BATCH) { sumE[i] = 0.0f; gsum[i] = 0.0f; }
}

// ---------------- kernel 1: row softmax of G -> bf16 Act^T [k][q] -----------
__global__ __launch_bounds__(256) void gsoftmax_kernel(
    const float* __restrict__ G, unsigned short* __restrict__ ActT) {
  int q = blockIdx.x;
  __shared__ float red[256];
  int tid = threadIdx.x;
  const float* row = G + (size_t)q * L;

  float mx = -3.4e38f;
  for (int k = tid; k < L; k += 256) mx = fmaxf(mx, row[k]);
  red[tid] = mx; __syncthreads();
  for (int off = 128; off; off >>= 1) {
    if (tid < off) red[tid] = fmaxf(red[tid], red[tid + off]);
    __syncthreads();
  }
  mx = red[0]; __syncthreads();

  float sm = 0.0f;
  for (int k = tid; k < L; k += 256) sm += __expf(row[k] - mx);
  red[tid] = sm; __syncthreads();
  for (int off = 128; off; off >>= 1) {
    if (tid < off) red[tid] += red[tid + off];
    __syncthreads();
  }
  float inv = 1.0f / red[0];

  for (int k = tid; k < L; k += 256)
    ActT[(size_t)k * L + q] = f2bf(__expf(row[k] - mx) * inv);  // transposed
}

// ---------------- kernel 2: QKV projection GEMM (xb @ W^T), WMMA bf16 -------
// grid = (MROWS/16, E/64, 3). One wave computes 16x64 (A reused by 4 WMMAs).
// Q,K head-major [bh][L][D]; V stored TRANSPOSED [bh][D][L] for P@V B-frags.
__global__ __launch_bounds__(32) void qkv_kernel(
    const unsigned short* __restrict__ xb, const unsigned short* __restrict__ Wb,
    unsigned short* __restrict__ qkv) {
  int lane = threadIdx.x, half = lane >> 4, lx = lane & 15;
  int m0 = blockIdx.x * 16, n0 = blockIdx.y * 64, mat = blockIdx.z;
  const unsigned short* W = Wb + (size_t)mat * E * E;          // row-major [n][k]
  unsigned short* outp = qkv + (size_t)mat * BH * L * D;

  const unsigned short* arow = xb + (size_t)(m0 + lx) * E;
  const unsigned short* brow0 = W + (size_t)(n0 + lx) * E;     // cols n0+lx+16*nt
  v8f acc[4] = {{}, {}, {}, {}};
  for (int kt = 0; kt < E; kt += 32) {
    v16u a = load_frag(arow + kt, half);
#pragma unroll
    for (int nt = 0; nt < 4; ++nt) {
      v16u bm = load_frag(brow0 + (size_t)nt * 16 * E + kt, half);
      acc[nt] = wmma_bf16(a, bm, acc[nt]);
    }
  }
#pragma unroll
  for (int nt = 0; nt < 4; ++nt)
#pragma unroll
    for (int r = 0; r < 8; ++r) {
      int m = m0 + r + half * 8, n = n0 + nt * 16 + lx;
      int b = m >> 10, l = m & 1023, h = n >> 6, d = n & 63;
      unsigned short val = f2bf(acc[nt][r]);
      if (mat == 2)   // V transposed: [bh][D][L]
        outp[(((size_t)(b * H + h)) * D + d) * L + l] = val;
      else
        outp[(((size_t)(b * H + h)) * L + l) * D + d] = val;
    }
}

// ---------------- kernel 3: global denominator sumE[b] ----------------------
// grid = (BH, L/16), one wave: recompute QK^T tiles, sum exp(tanh)*masks.
__global__ __launch_bounds__(32) void sumE_kernel(
    const unsigned short* __restrict__ Qb, const unsigned short* __restrict__ Kb,
    const float* __restrict__ maskf, float* __restrict__ sumE) {
  int lane = threadIdx.x, half = lane >> 4, lx = lane & 15;
  int bh = blockIdx.x, b = bh >> 3;
  int q0 = blockIdx.y * 16;
  const unsigned short* Qh = Qb + (size_t)bh * L * D;
  const unsigned short* Kh = Kb + (size_t)bh * L * D;
  const float* mrow = maskf + (size_t)b * L;

  const unsigned short* qrow = Qh + (size_t)(q0 + lx) * D;
  v16u a0 = load_frag(qrow, half);
  v16u a1 = load_frag(qrow + 32, half);
  float mq[8];
#pragma unroll
  for (int r = 0; r < 8; ++r) mq[r] = mrow[q0 + r + half * 8];

  float lsum = 0.0f;
  for (int n0 = 0; n0 < L; n0 += 16) {
    const unsigned short* krow = Kh + (size_t)(n0 + lx) * D;
    v16u b0 = load_frag(krow, half);
    v16u b1 = load_frag(krow + 32, half);
    v8f s = {};
    s = wmma_bf16(a0, b0, s);
    s = wmma_bf16(a1, b1, s);
    float mk = mrow[n0 + lx];
    if (mk != 0.0f) {
      float t = 0.0f;
#pragma unroll
      for (int r = 0; r < 8; ++r)
        t += __expf(tanhf(s[r] * SCALE)) * mq[r];
      lsum += t;   // * mk == 1
    }
  }
#pragma unroll
  for (int off = 16; off; off >>= 1) lsum += __shfl_xor(lsum, off, 32);
  if (lane == 0) atomicAdd(&sumE[b], lsum);
}

// ---------------- kernel 4: gsum[b] = sum_k mk * sum_q actT[k,q]*mq ---------
__global__ __launch_bounds__(256) void gsum_kernel(
    const unsigned short* __restrict__ ActT, const float* __restrict__ maskf,
    float* __restrict__ gsum) {
  int b = blockIdx.x, k0 = blockIdx.y * 8;
  const float* mrow = maskf + (size_t)b * L;
  __shared__ float red[256];
  float acc = 0.0f;
  for (int kk = 0; kk < 8; ++kk) {
    int k = k0 + kk;
    float mk = mrow[k];
    if (mk != 0.0f) {
      float s = 0.0f;
      for (int q = threadIdx.x; q < L; q += 256)
        s += bf2f(ActT[(size_t)k * L + q]) * mrow[q];
      acc += mk * s;
    }
  }
  red[threadIdx.x] = acc; __syncthreads();
  for (int off = 128; off; off >>= 1) {
    if (threadIdx.x < off) red[threadIdx.x] += red[threadIdx.x + off];
    __syncthreads();
  }
  if (threadIdx.x == 0) atomicAdd(&gsum[b], red[0]);
}

// ---------------- kernel 5: flash-style attention, P = c1*E' + c2*Act -------
// grid = (BH, L/16). Stage P tile through LDS to re-layout C->A fragment.
// Output row-major [B*L][E] (K-contiguous for the Wo GEMM).
__global__ __launch_bounds__(32) void attn_kernel(
    const unsigned short* __restrict__ Qb, const unsigned short* __restrict__ Kb,
    const unsigned short* __restrict__ Vt, const unsigned short* __restrict__ ActT,
    const float* __restrict__ maskf, const float* __restrict__ sumE,
    const float* __restrict__ gsum, unsigned short* __restrict__ Att) {
  int lane = threadIdx.x, half = lane >> 4, lx = lane & 15;
  int bh = blockIdx.x, b = bh >> 3, h = bh & 7;
  int q0 = blockIdx.y * 16;
  const unsigned short* Qh = Qb + (size_t)bh * L * D;
  const unsigned short* Kh = Kb + (size_t)bh * L * D;
  const unsigned short* Vh = Vt + (size_t)bh * D * L;   // transposed [D][L]
  const float* mrow = maskf + (size_t)b * L;
  float c1 = 0.5f / fmaxf(sumE[b], 1e-30f);
  float c2 = 0.5f / fmaxf(gsum[b], 1e-30f);

  __shared__ alignas(16) unsigned short pt[16][32];  // 16x32 bf16 P tile

  const unsigned short* qrow = Qh + (size_t)(q0 + lx) * D;
  v16u a0 = load_frag(qrow, half);
  v16u a1 = load_frag(qrow + 32, half);
  v8f acc[4] = {{}, {}, {}, {}};

  for (int k0 = 0; k0 < L; k0 += 32) {
#pragma unroll
    for (int t = 0; t < 2; ++t) {
      int n0 = k0 + t * 16;
      const unsigned short* krow = Kh + (size_t)(n0 + lx) * D;
      v16u b0 = load_frag(krow, half);
      v16u b1 = load_frag(krow + 32, half);
      v8f s = {};
      s = wmma_bf16(a0, b0, s);
      s = wmma_bf16(a1, b1, s);
      int k = n0 + lx;
      float mk = mrow[k];
      // 8 act values for this lane's column: one b128 load (ActT is [k][q])
      u16x8 av = *(const u16x8*)(ActT + (size_t)k * L + q0 + half * 8);
#pragma unroll
      for (int r = 0; r < 8; ++r) {
        float e = __expf(tanhf(s[r] * SCALE));
        float p = (c1 * e + c2 * bf2f(av[r])) * mk;
        pt[r + half * 8][t * 16 + lx] = f2bf(p);
      }
    }
    __syncthreads();
    v16u pa = load_frag(&pt[lx][0], half);   // two ds_load_b128
#pragma unroll
    for (int dt = 0; dt < 4; ++dt) {
      v16u vb = load_frag(Vh + (size_t)(dt * 16 + lx) * L + k0, half);
      acc[dt] = wmma_bf16(pa, vb, acc[dt]);
    }
    __syncthreads();
  }
#pragma unroll
  for (int dt = 0; dt < 4; ++dt)
#pragma unroll
    for (int r = 0; r < 8; ++r) {
      int q = q0 + r + half * 8;
      Att[((size_t)(b * L + q)) * E + h * 64 + dt * 16 + lx] =
          f2bf(acc[dt][r] * mrow[q]);
    }
}

// ---------------- kernel 6: output projection attended @ Wo^T ---------------
// grid = (MROWS/16, E/64); A is row-major [M][E] bf16, 16x64 per wave.
__global__ __launch_bounds__(32) void out_kernel(
    const unsigned short* __restrict__ Att, const unsigned short* __restrict__ Wob,
    float* __restrict__ out) {
  int lane = threadIdx.x, half = lane >> 4, lx = lane & 15;
  int m0 = blockIdx.x * 16, n0 = blockIdx.y * 64;
  const unsigned short* arow = Att + (size_t)(m0 + lx) * E;
  const unsigned short* brow0 = Wob + (size_t)(n0 + lx) * E;
  v8f acc[4] = {{}, {}, {}, {}};
  for (int kt = 0; kt < E; kt += 32) {
    v16u a = load_frag(arow + kt, half);
#pragma unroll
    for (int nt = 0; nt < 4; ++nt) {
      v16u bm = load_frag(brow0 + (size_t)nt * 16 * E + kt, half);
      acc[nt] = wmma_bf16(a, bm, acc[nt]);
    }
  }
#pragma unroll
  for (int nt = 0; nt < 4; ++nt)
#pragma unroll
    for (int r = 0; r < 8; ++r) {
      int m = m0 + r + half * 8, n = n0 + nt * 16 + lx;
      out[(size_t)m * E + n] = acc[nt][r];
    }
}

// ---------------- host launcher ----------------
extern "C" void kernel_launch(void* const* d_in, const int* in_sizes, int n_in,
                              void* d_out, int out_size, void* d_ws, size_t ws_size,
                              hipStream_t stream) {
  const float* x    = (const float*)d_in[0];
  const int*   mask = (const int*)d_in[1];
  const float* Wq   = (const float*)d_in[2];
  const float* Wk   = (const float*)d_in[3];
  const float* Wv   = (const float*)d_in[4];
  const float* Wo   = (const float*)d_in[5];
  const float* G    = (const float*)d_in[6];
  float* out = (float*)d_out;

  char* base = (char*)d_ws;
  size_t off = 0;
  auto take = [&](size_t bytes) -> void* {
    void* p = base + off;
    off = (off + bytes + 255) & ~(size_t)255;
    return p;
  };
  const size_t HEADN = (size_t)BH * L * D;                     // elements
  unsigned short* xb   = (unsigned short*)take((size_t)MROWS * E * 2);
  unsigned short* qkv  = (unsigned short*)take(3 * HEADN * 2); // Q,K,Vt bf16
  unsigned short* Qb   = qkv;
  unsigned short* Kb   = qkv + HEADN;
  unsigned short* Vt   = qkv + 2 * HEADN;
  unsigned short* Att  = (unsigned short*)take((size_t)MROWS * E * 2);
  unsigned short* ActT = (unsigned short*)take((size_t)L * L * 2);
  unsigned short* Wb   = (unsigned short*)take((size_t)4 * E * E * 2);
  float* maskf = (float*)take((size_t)BATCH * L * 4);
  float* sumE  = (float*)take(BATCH * 4);
  float* gsum  = (float*)take(BATCH * 4);

  const int XN4 = MROWS * E / 4;
  prep_kernel<<<dim3((XN4 + 255) / 256), dim3(256), 0, stream>>>(
      x, xb, Wq, Wk, Wv, Wo, Wb, mask, maskf, sumE, gsum);
  gsoftmax_kernel<<<dim3(L), dim3(256), 0, stream>>>(G, ActT);
  qkv_kernel<<<dim3(MROWS / 16, E / 64, 3), dim3(32), 0, stream>>>(xb, Wb, qkv);
  sumE_kernel<<<dim3(BH, L / 16), dim3(32), 0, stream>>>(Qb, Kb, maskf, sumE);
  gsum_kernel<<<dim3(BATCH, L / 8), dim3(256), 0, stream>>>(ActT, maskf, gsum);
  attn_kernel<<<dim3(BH, L / 16), dim3(32), 0, stream>>>(
      Qb, Kb, Vt, ActT, maskf, sumE, gsum, Att);
  out_kernel<<<dim3(MROWS / 16, E / 64), dim3(32), 0, stream>>>(
      Att, Wb + 3 * (size_t)E * E, out);
}
